// SAGE_35330400977011
// MI455X (gfx1250) — matrix operand order repaired and checked by reading, not compiled, not run on hardware
//
#include <hip/hip_runtime.h>
#include <hip/hip_bf16.h>

typedef __attribute__((ext_vector_type(16))) __bf16 v16bf;
typedef __attribute__((ext_vector_type(8)))  float  v8f;

#define KDIM 128   // F_IN == H == 128

// ---------- helpers ----------

// native CDNA5 float atomic add (no return -> STOREcnt path)
__device__ __forceinline__ void atomic_fadd(float* p, float v) {
    asm volatile("global_atomic_add_f32 %0, %1, off" :: "v"(p), "v"(v) : "memory");
}

// pack 4 floats into bf16 vector positions [base, base+3] (base is compile-time)
__device__ __forceinline__ void put4(v16bf& v, int base, float4 f) {
    v[base + 0] = (__bf16)f.x;
    v[base + 1] = (__bf16)f.y;
    v[base + 2] = (__bf16)f.z;
    v[base + 3] = (__bf16)f.w;
}

// per-chunk pipeline hint: 8 vmem reads -> 16 valu (cvt) -> 2 wmma
#if defined(__has_builtin) && __has_builtin(__builtin_amdgcn_sched_group_barrier)
#define CHUNK_SCHED()                                         \
    __builtin_amdgcn_sched_group_barrier(0x020, 8, 0);  /* VMEM_READ */ \
    __builtin_amdgcn_sched_group_barrier(0x002, 16, 0); /* VALU      */ \
    __builtin_amdgcn_sched_group_barrier(0x008, 2, 0)   /* MFMA/WMMA */
#else
#define CHUNK_SCHED()
#endif

// ---------- kernels ----------

__global__ void sage_zero(float* __restrict__ p, long n) {
    long i = (long)blockIdx.x * blockDim.x + threadIdx.x;
    if (i < n) p[i] = 0.0f;
}

__global__ void sage_degree(const int* __restrict__ dst, float* __restrict__ deg, int E) {
    int i = blockIdx.x * blockDim.x + threadIdx.x;
    if (i < E) atomic_fadd(&deg[dst[i]], 1.0f);
}

// deg[i] <- 1 / max(deg[i], 1)   (so the gemm epilogue is a pure load+fma)
__global__ void sage_invdeg(float* __restrict__ deg, int N) {
    int i = blockIdx.x * blockDim.x + threadIdx.x;
    if (i < N) deg[i] = 1.0f / fmaxf(deg[i], 1.0f);
}

// one wave32 per edge: lane l moves floats [4l, 4l+3] of the 128-float row
__global__ void sage_scatter(const float* __restrict__ h,
                             const int*  __restrict__ src,
                             const int*  __restrict__ dst,
                             float* __restrict__ agg, int E) {
    int edge = blockIdx.x * (blockDim.x >> 5) + (threadIdx.x >> 5);
    int lane = threadIdx.x & 31;
    if (edge >= E) return;
    int s = src[edge];
    int d = dst[edge];
    const float4* hrow = (const float4*)(h + (long)s * KDIM);
    float4 v = hrow[lane];
    float* arow = agg + (long)d * KDIM + lane * 4;
    atomic_fadd(arow + 0, v.x);
    atomic_fadd(arow + 1, v.y);
    atomic_fadd(arow + 2, v.z);
    atomic_fadd(arow + 3, v.w);
}

// fused: out = act( h @ Wself + inv[row] * (agg @ Wneigh) + b )
// Split accumulators: accS = h@Ws, accM = agg@Wn; the per-row 1/deg scale is
// applied once in the epilogue (distributes over the K reduction), so both A
// streams are pure load -> v_cvt_pk_bf16 -> wmma.
template <bool RELU>
__global__ void sage_gemm(const float* __restrict__ h,      // [N,128]
                          const float* __restrict__ agg,    // [N,128]
                          const float* __restrict__ invd,   // [N] = 1/max(deg,1)
                          const float* __restrict__ Wself,  // [128,Dout]
                          const float* __restrict__ Wneigh, // [128,Dout]
                          const float* __restrict__ bias,   // [Dout]
                          float* __restrict__ out,          // [N,Dout]
                          int N, int Dout, int tilesN, int rowGroups) {
    constexpr int MT = 8;                         // row tiles per wave
    int tid = blockIdx.x * (blockDim.x >> 5) + (threadIdx.x >> 5);
    if (tid >= tilesN * rowGroups) return;        // whole-wave exit: EXEC all-1s

    int rg = tid / tilesN;
    int tn = tid - rg * tilesN;
    int n0 = tn << 4;

    int lane = threadIdx.x & 31;
    int half = lane >> 4;      // 0: lanes 0-15, 1: lanes 16-31
    int sub  = lane & 15;

    int  gcol  = n0 + sub;
    bool colok = gcol < Dout;
    int  ccol  = colok ? gcol : (Dout - 1);  // clamped: junk cols only feed unstored lanes

    // ---- load ALL of B once: 4 K-chunks x {Wself, Wneigh}, unpredicated ----
    v16bf bS[4], bM[4];
#pragma unroll
    for (int c = 0; c < 4; ++c) {
#pragma unroll
        for (int e = 0; e < 16; ++e) {
            // 16-bit B 32x16 layout: e<8 -> K = e + half*8, e>=8 -> K = 16+(e-8) + half*8
            int kb = (e < 8 ? e : e + 8) + half * 8 + c * 32;
            bS[c][e] = (__bf16)Wself [(long)kb * Dout + ccol];
            bM[c][e] = (__bf16)Wneigh[(long)kb * Dout + ccol];
        }
    }
    float bv = bias[ccol];

    int tilesM = (N + 15) >> 4;
    int tmBase = rg * MT;

#pragma unroll 1
    for (int i = 0; i < MT; ++i) {
        int tm = tmBase + i;
        if (tm >= tilesM) break;                  // uniform across wave
        int m0 = tm << 4;

        int arow = m0 + sub;
        if (arow > N - 1) arow = N - 1;
        // pre-offset by half*8: lane's two runs per chunk sit at float4 idx {0,1} and {4,5}
        const float4* hrow = (const float4*)(h   + (long)arow * KDIM + half * 8);
        const float4* grow = (const float4*)(agg + (long)arow * KDIM + half * 8);

        v8f accS, accM;
#pragma unroll
        for (int r = 0; r < 8; ++r) { accS[r] = bv; accM[r] = 0.0f; }

#pragma unroll
        for (int c = 0; c < 4; ++c) {
            // ---- 8 independent 128-bit loads, then convert, then 2 WMMAs ----
            float4 h0 = hrow[c * 8 + 0];
            float4 h1 = hrow[c * 8 + 1];
            float4 h2 = hrow[c * 8 + 4];
            float4 h3 = hrow[c * 8 + 5];
            float4 g0 = grow[c * 8 + 0];
            float4 g1 = grow[c * 8 + 1];
            float4 g2 = grow[c * 8 + 4];
            float4 g3 = grow[c * 8 + 5];

            v16bf aS, aM;
            // 16-bit A 16x32 layout (ISA 7.12.2): e<8 -> K = e (+half*8+c*32),
            // e>=8 -> K = 16+(e-8) (+half*8+c*32)
            put4(aS, 0, h0);  put4(aS, 4, h1);  put4(aS, 8, h2);  put4(aS, 12, h3);
            put4(aM, 0, g0);  put4(aM, 4, g1);  put4(aM, 8, g2);  put4(aM, 12, g3);

            accS = __builtin_amdgcn_wmma_f32_16x16x32_bf16(false, aS, false, bS[c],
                                                           (short)0, accS, false, false);
            accM = __builtin_amdgcn_wmma_f32_16x16x32_bf16(false, aM, false, bM[c],
                                                           (short)0, accM, false, false);
            CHUNK_SCHED();
        }

        if (colok) {
#pragma unroll
            for (int r = 0; r < 8; ++r) {
                int orow = m0 + r + half * 8;     // C/D layout: M = vgpr + 8*(lane>=16)
                if (orow < N) {
                    float v = accS[r] + invd[orow] * accM[r];
                    if (RELU) v = fmaxf(v, 0.0f);
                    out[(long)orow * Dout + gcol] = v;
                }
            }
        }
    }
}

// ---------- launcher ----------

extern "C" void kernel_launch(void* const* d_in, const int* in_sizes, int n_in,
                              void* d_out, int out_size, void* d_ws, size_t ws_size,
                              hipStream_t stream) {
    const float* x        = (const float*)d_in[0];
    const int*   src      = (const int*)  d_in[1];
    const int*   dst      = (const int*)  d_in[2];
    const float* w_self0  = (const float*)d_in[3];
    const float* w_neigh0 = (const float*)d_in[4];
    const float* b0       = (const float*)d_in[5];
    const float* w_self1  = (const float*)d_in[6];
    const float* w_neigh1 = (const float*)d_in[7];
    const float* b1       = (const float*)d_in[8];
    const float* w_self2  = (const float*)d_in[9];
    const float* w_neigh2 = (const float*)d_in[10];
    const float* b2       = (const float*)d_in[11];

    const int N = in_sizes[0] / KDIM;   // 100000
    const int E = in_sizes[1];          // 1600000
    const int C = in_sizes[11];         // 47

    float* agg = (float*)d_ws;
    float* deg = agg + (long)N * KDIM;  // becomes 1/max(deg,1) after sage_invdeg
    float* h1  = deg + N;
    float* h2  = h1  + (long)N * KDIM;

    float* outp = (float*)d_out;

    const long aggN   = (long)N * KDIM;
    const int  zblk   = 256;
    const int  sWaves = 8;                                // waves per scatter block (256 thr)
    const int  sGrid  = (E + sWaves - 1) / sWaves;
    const int  gWaves = 4;                                // waves per gemm block (128 thr)
    const int  MT     = 8;                                // row tiles per wave (match kernel)

    int tilesM    = (N + 15) / 16;
    int rowGroups = (tilesM + MT - 1) / MT;
    int tilesN_h  = (KDIM + 15) / 16;                     // 8
    int tilesN_c  = (C + 15) / 16;                        // 3
    int jobsH = tilesN_h * rowGroups;
    int jobsC = tilesN_c * rowGroups;
    int gGridH = (jobsH + gWaves - 1) / gWaves;
    int gGridC = (jobsC + gWaves - 1) / gWaves;

    // ---- degree -> 1/max(deg,1) (constant across layers, recomputed per call) ----
    sage_zero<<<(int)((aggN + N + zblk - 1) / zblk), zblk, 0, stream>>>(agg, aggN + N);
    sage_degree<<<(E + zblk - 1) / zblk, zblk, 0, stream>>>(dst, deg, E);
    sage_invdeg<<<(N + zblk - 1) / zblk, zblk, 0, stream>>>(deg, N);

    // ---- layer 0 ----
    sage_scatter<<<sGrid, 256, 0, stream>>>(x, src, dst, agg, E);
    sage_gemm<true><<<gGridH, 128, 0, stream>>>(x, agg, deg, w_self0, w_neigh0, b0,
                                                h1, N, KDIM, tilesN_h, rowGroups);
    // ---- layer 1 ----
    sage_zero<<<(int)((aggN + zblk - 1) / zblk), zblk, 0, stream>>>(agg, aggN);
    sage_scatter<<<sGrid, 256, 0, stream>>>(h1, src, dst, agg, E);
    sage_gemm<true><<<gGridH, 128, 0, stream>>>(h1, agg, deg, w_self1, w_neigh1, b1,
                                                h2, N, KDIM, tilesN_h, rowGroups);
    // ---- layer 2 (no relu, Dout=47) ----
    sage_zero<<<(int)((aggN + zblk - 1) / zblk), zblk, 0, stream>>>(agg, aggN);
    sage_scatter<<<sGrid, 256, 0, stream>>>(h2, src, dst, agg, E);
    sage_gemm<false><<<gGridC, 128, 0, stream>>>(h2, agg, deg, w_self2, w_neigh2, b2,
                                                 outp, N, C, tilesN_c, rowGroups);
}
